// Dual_GCN_60610578481666
// MI455X (gfx1250) — matrix operand order repaired and checked by reading, not compiled, not run on hardware
//
#include <hip/hip_runtime.h>
#include <math.h>

// ---------------------------------------------------------------------------
// Types for CDNA5 WMMA (gfx1250, wave32)
// ---------------------------------------------------------------------------
typedef __attribute__((ext_vector_type(16))) __bf16       v16bf;
typedef __attribute__((ext_vector_type(8)))  float        v8f;
typedef __attribute__((ext_vector_type(4)))  unsigned int v4u;

#define D_FEAT 128
#define HATT   64

union ABFrag { v4u q[2]; v16bf v; };

__device__ __forceinline__ float waveReduceSum(float v) {
#pragma unroll
  for (int off = 16; off > 0; off >>= 1) v += __shfl_xor(v, off, 32);
  return v;
}

// ---------------------------------------------------------------------------
// Elementwise helpers
// ---------------------------------------------------------------------------
__global__ void fill_kernel(float* __restrict__ p, float v, int n) {
  int i = blockIdx.x * blockDim.x + threadIdx.x;
  if (i < n) p[i] = v;
}

__global__ void convert_bf16_kernel(const float* __restrict__ src,
                                    __bf16* __restrict__ dst, int n) {
  int i = blockIdx.x * blockDim.x + threadIdx.x;
  if (i < n) dst[i] = (__bf16)src[i];
}

// Wt[n*128 + k] = (bf16) W[k*128 + n]   (column-major B for WMMA K-contig loads)
__global__ void transpose_convert_kernel(const float* __restrict__ W,
                                         __bf16* __restrict__ Wt) {
  int i = blockIdx.x * blockDim.x + threadIdx.x;  // i < 128*128
  int k = i >> 7, n = i & 127;
  Wt[n * D_FEAT + k] = (__bf16)W[k * D_FEAT + n];
}

__global__ void relu_kernel(float* __restrict__ p, int n) {
  int i = blockIdx.x * blockDim.x + threadIdx.x;
  if (i < n) p[i] = fmaxf(p[i], 0.0f);
}

__global__ void dis_kernel(const float* __restrict__ deg,
                           float* __restrict__ dis, int n) {
  int i = blockIdx.x * blockDim.x + threadIdx.x;
  if (i < n) {
    float d = deg[i];
    dis[i] = (d > 0.0f) ? rsqrtf(fmaxf(d, 1e-12f)) : 0.0f;
  }
}

// ---------------------------------------------------------------------------
// Edge weight kernels: one wave per edge, one float4 per lane (= full 128 dim)
// ---------------------------------------------------------------------------
__global__ void edge_cos_kernel(const float* __restrict__ x,
                                const int* __restrict__ row,
                                const int* __restrict__ col,
                                float* __restrict__ ew,
                                float* __restrict__ deg, int E) {
  const int lane = threadIdx.x & 31;
  const int e = blockIdx.x * 8 + (threadIdx.x >> 5);
  if (e >= E) return;
  const int r = row[e], c = col[e];
  const float4 a = reinterpret_cast<const float4*>(x + (size_t)r * D_FEAT)[lane];
  const float4 b = reinterpret_cast<const float4*>(x + (size_t)c * D_FEAT)[lane];
  float dot = a.x * b.x + a.y * b.y + a.z * b.z + a.w * b.w;
  float na  = a.x * a.x + a.y * a.y + a.z * a.z + a.w * a.w;
  float nb  = b.x * b.x + b.y * b.y + b.z * b.z + b.w * b.w;
  dot = waveReduceSum(dot);
  na  = waveReduceSum(na);
  nb  = waveReduceSum(nb);
  if (lane == 0) {
    float w = dot / fmaxf(sqrtf(na) * sqrtf(nb), 1e-8f);
    ew[e] = w;
    atomicAdd(deg + r, w);
  }
}

__global__ void edge_euc_kernel(const float* __restrict__ x,
                                const int* __restrict__ row,
                                const int* __restrict__ col,
                                float* __restrict__ ew,
                                float* __restrict__ deg, int E) {
  const int lane = threadIdx.x & 31;
  const int e = blockIdx.x * 8 + (threadIdx.x >> 5);
  if (e >= E) return;
  const int r = row[e], c = col[e];
  const float4 a = reinterpret_cast<const float4*>(x + (size_t)r * D_FEAT)[lane];
  const float4 b = reinterpret_cast<const float4*>(x + (size_t)c * D_FEAT)[lane];
  float dx = a.x - b.x, dy = a.y - b.y, dz = a.z - b.z, dw = a.w - b.w;
  float s = dx * dx + dy * dy + dz * dz + dw * dw;
  s = waveReduceSum(s);
  if (lane == 0) {
    float w = sqrtf(s + 1e-12f);
    ew[e] = w;
    atomicAdd(deg + r, w);
  }
}

// ---------------------------------------------------------------------------
// WMMA GEMM:  C[M x 128] = A[M x 128] (bf16, row-major) * W  (via Bt[n][k])
//   Each wave computes TWO 16x16 output tiles (rows m0 and m0+16) against the
//   same B fragment -> B loads halved, 8 WMMAs per wave. 8 waves/block cover
//   all 128 output columns. Stores use a uniform full-tile fast path.
// ---------------------------------------------------------------------------
__global__ void gemm_bf16_wmma(const __bf16* __restrict__ A,
                               const __bf16* __restrict__ Bt,
                               float* __restrict__ C, int M) {
  const int lane = threadIdx.x & 31;
  const int wave = threadIdx.x >> 5;
  const int m0 = blockIdx.x * 32;      // two 16-row tiles: m0, m0+16
  const int n0 = wave * 16;
  const int half = lane >> 4;          // 0: lanes 0-15, 1: lanes 16-31
  const int lr = lane & 15;

  int ar0 = m0 + lr;      if (ar0 >= M) ar0 = M - 1;   // clamp loads only
  int ar1 = m0 + 16 + lr; if (ar1 >= M) ar1 = M - 1;
  const __bf16* arow0 = A  + (size_t)ar0 * D_FEAT;
  const __bf16* arow1 = A  + (size_t)ar1 * D_FEAT;
  const __bf16* brow  = Bt + (size_t)(n0 + lr) * D_FEAT;

  v8f acc0 = {0.f, 0.f, 0.f, 0.f, 0.f, 0.f, 0.f, 0.f};
  v8f acc1 = {0.f, 0.f, 0.f, 0.f, 0.f, 0.f, 0.f, 0.f};
#pragma unroll
  for (int k0 = 0; k0 < D_FEAT; k0 += 32) {
    ABFrag a0, a1, b;
    // A 16x32 bf16 layout: lanes 0-15 hold K {0..7,16..23}, lanes 16-31 {8..15,24..31}
    const int ka = k0 + half * 8;
    a0.q[0] = *reinterpret_cast<const v4u*>(arow0 + ka);
    a0.q[1] = *reinterpret_cast<const v4u*>(arow0 + ka + 16);
    a1.q[0] = *reinterpret_cast<const v4u*>(arow1 + ka);
    a1.q[1] = *reinterpret_cast<const v4u*>(arow1 + ka + 16);
    // B 32x16 bf16 layout: lanes 0-15 hold K {0..7,16..23}, lanes 16-31 {8..15,24..31}
    const int kb = k0 + half * 8;
    b.q[0] = *reinterpret_cast<const v4u*>(brow + kb);
    b.q[1] = *reinterpret_cast<const v4u*>(brow + kb + 16);
    acc0 = __builtin_amdgcn_wmma_f32_16x16x32_bf16(
        false, a0.v, false, b.v, (short)0, acc0, false, false);
    acc1 = __builtin_amdgcn_wmma_f32_16x16x32_bf16(
        false, a1.v, false, b.v, (short)0, acc1, false, false);
  }
  // C 16x16 f32 layout: VGPR j -> M = tile + j + half*8, N = n0 + (lane&15)
  const int colc = n0 + lr;
  if (m0 + 32 <= M) {  // uniform fast path: both tiles fully in range
#pragma unroll
    for (int j = 0; j < 8; ++j) {
      C[(size_t)(m0 + j + half * 8) * D_FEAT + colc]      = acc0[j];
      C[(size_t)(m0 + 16 + j + half * 8) * D_FEAT + colc] = acc1[j];
    }
  } else {
#pragma unroll
    for (int j = 0; j < 8; ++j) {
      const int ma = m0 + j + half * 8;
      const int mb = ma + 16;
      if (ma < M) C[(size_t)ma * D_FEAT + colc] = acc0[j];
      if (mb < M) C[(size_t)mb * D_FEAT + colc] = acc1[j];
    }
  }
}

// ---------------------------------------------------------------------------
// out[i,f] = dis[i]^2 * h[i,f] + bias[f]   (self-loop term + bias, full init)
// ---------------------------------------------------------------------------
__global__ void selfloop_bias_kernel(const float* __restrict__ h,
                                     const float* __restrict__ dis,
                                     const float* __restrict__ bias,
                                     float* __restrict__ out, int total) {
  int i = blockIdx.x * blockDim.x + threadIdx.x;
  if (i >= total) return;
  int node = i >> 7, f = i & 127;
  float d = dis[node];
  out[i] = d * d * h[i] + bias[f];
}

// ---------------------------------------------------------------------------
// SpMM edge scatter: out[r] += dis[r]*w_e*dis[c] * h[c]; wave per edge
// ---------------------------------------------------------------------------
__global__ void spmm_edge_kernel(const float* __restrict__ h,
                                 const float* __restrict__ ew,
                                 const int* __restrict__ row,
                                 const int* __restrict__ col,
                                 const float* __restrict__ dis,
                                 float* __restrict__ out, int E) {
  const int lane = threadIdx.x & 31;
  const int e = blockIdx.x * 8 + (threadIdx.x >> 5);
  if (e >= E) return;
  const int r = row[e], c = col[e];
  if (e + 8 < E)  // pull the next edge's source row toward L2/L0
    __builtin_prefetch(h + (size_t)col[e + 8] * D_FEAT, 0, 3);
  const float norm = dis[r] * ew[e] * dis[c];
  const float4 hv = reinterpret_cast<const float4*>(h + (size_t)c * D_FEAT)[lane];
  float* o = out + (size_t)r * D_FEAT + lane * 4;
  atomicAdd(o + 0, norm * hv.x);
  atomicAdd(o + 1, norm * hv.y);
  atomicAdd(o + 2, norm * hv.z);
  atomicAdd(o + 3, norm * hv.w);
}

// ---------------------------------------------------------------------------
// Semantic attention fusion: one wave per node, embeddings staged in LDS
// ---------------------------------------------------------------------------
__global__ void fusion_kernel(const float* __restrict__ x1,
                              const float* __restrict__ x2,
                              const float* __restrict__ A1,   // [128,64]
                              const float* __restrict__ ab1,  // [64]
                              const float* __restrict__ A2,   // [64]
                              float* __restrict__ out, int N) {
  __shared__ float sx[8][2][D_FEAT];
  const int lane = threadIdx.x & 31;
  const int wave = threadIdx.x >> 5;
  const int node = blockIdx.x * 8 + wave;
  const int nn = node < N ? node : N - 1;
  const float4 v1 = reinterpret_cast<const float4*>(x1 + (size_t)nn * D_FEAT)[lane];
  const float4 v2 = reinterpret_cast<const float4*>(x2 + (size_t)nn * D_FEAT)[lane];
  *reinterpret_cast<float4*>(&sx[wave][0][lane * 4]) = v1;
  *reinterpret_cast<float4*>(&sx[wave][1][lane * 4]) = v2;
  __syncthreads();

  float s1 = 0.f, s2 = 0.f;
#pragma unroll
  for (int jj = 0; jj < 2; ++jj) {
    const int j = lane + jj * 32;
    float a1 = ab1[j], a2 = ab1[j];
    for (int d = 0; d < D_FEAT; ++d) {
      const float w = A1[d * HATT + j];
      a1 += sx[wave][0][d] * w;
      a2 += sx[wave][1][d] * w;
    }
    const float av = A2[j];
    s1 += tanhf(a1) * av;
    s2 += tanhf(a2) * av;
  }
  s1 = waveReduceSum(s1);
  s2 = waveReduceSum(s2);
  const float m = fmaxf(s1, s2);
  const float e1 = __expf(s1 - m), e2 = __expf(s2 - m);
  const float inv = 1.0f / (e1 + e2);
  const float be1 = e1 * inv, be2 = e2 * inv;
  if (node < N) {
    float4 o;
    o.x = be1 * v1.x + be2 * v2.x;
    o.y = be1 * v1.y + be2 * v2.y;
    o.z = be1 * v1.z + be2 * v2.z;
    o.w = be1 * v1.w + be2 * v2.w;
    *reinterpret_cast<float4*>(out + (size_t)node * D_FEAT + lane * 4) = o;
  }
}

// ---------------------------------------------------------------------------
// Host orchestration
// ---------------------------------------------------------------------------
extern "C" void kernel_launch(void* const* d_in, const int* in_sizes, int n_in,
                              void* d_out, int out_size, void* d_ws, size_t ws_size,
                              hipStream_t stream) {
  const float* x   = (const float*)d_in[0];
  const int*   row = (const int*)d_in[1];
  const int*   col = (const int*)d_in[2];
  const float* W1  = (const float*)d_in[3];
  const float* b1  = (const float*)d_in[4];
  const float* W2  = (const float*)d_in[5];
  const float* b2  = (const float*)d_in[6];
  const float* A1  = (const float*)d_in[7];
  const float* ab1 = (const float*)d_in[8];
  const float* A2  = (const float*)d_in[9];
  float* out = (float*)d_out;

  const int N = in_sizes[0] / D_FEAT;
  const int E = in_sizes[1];
  const int ND = N * D_FEAT;

  char* ws = (char*)d_ws;
  size_t off = 0;
  auto alloc = [&](size_t bytes) -> void* {
    void* p = ws + off;
    off += (bytes + 255) & ~(size_t)255;
    return p;
  };
  __bf16* xb  = (__bf16*)alloc((size_t)ND * 2);
  __bf16* fb  = (__bf16*)alloc((size_t)ND * 2);
  __bf16* W1t = (__bf16*)alloc((size_t)D_FEAT * D_FEAT * 2);
  __bf16* W2t = (__bf16*)alloc((size_t)D_FEAT * D_FEAT * 2);
  float* h1  = (float*)alloc((size_t)ND * 4);
  float* h2  = (float*)alloc((size_t)ND * 4);
  float* ew  = (float*)alloc((size_t)E * 4);
  float* deg = (float*)alloc((size_t)N * 4);
  float* dis = (float*)alloc((size_t)N * 4);
  float* c1  = (float*)alloc((size_t)ND * 4);
  float* c2  = (float*)alloc((size_t)ND * 4);
  float* x1o = (float*)alloc((size_t)ND * 4);
  float* x2o = (float*)alloc((size_t)ND * 4);

  const int B = 256;
  const int gND = (ND + B - 1) / B;
  const int gN  = (N + B - 1) / B;
  const int gE  = (E + 7) / 8;
  const int gM  = (N + 31) / 32;   // 32 rows per block (2 tiles per wave)
  const int gF  = (N + 7) / 8;
  const int gW  = (D_FEAT * D_FEAT + B - 1) / B;

  // Shared preprocessing
  convert_bf16_kernel<<<gND, B, 0, stream>>>(x, xb, ND);
  transpose_convert_kernel<<<gW, B, 0, stream>>>(W1, W1t);
  transpose_convert_kernel<<<gW, B, 0, stream>>>(W2, W2t);
  gemm_bf16_wmma<<<gM, B, 0, stream>>>(xb, W1t, h1, N);   // h1 = x@W1 (shared)

  // ---- cosine branch, layer 1 ----
  fill_kernel<<<gN, B, 0, stream>>>(deg, 1.0f, N);
  edge_cos_kernel<<<gE, B, 0, stream>>>(x, row, col, ew, deg, E);
  dis_kernel<<<gN, B, 0, stream>>>(deg, dis, N);
  selfloop_bias_kernel<<<gND, B, 0, stream>>>(h1, dis, b1, c1, ND);
  spmm_edge_kernel<<<gE, B, 0, stream>>>(h1, ew, row, col, dis, c1, E);
  relu_kernel<<<gND, B, 0, stream>>>(c1, ND);
  // ---- cosine branch, layer 2 ----
  fill_kernel<<<gN, B, 0, stream>>>(deg, 1.0f, N);
  edge_cos_kernel<<<gE, B, 0, stream>>>(c1, row, col, ew, deg, E);
  dis_kernel<<<gN, B, 0, stream>>>(deg, dis, N);
  convert_bf16_kernel<<<gND, B, 0, stream>>>(c1, fb, ND);
  gemm_bf16_wmma<<<gM, B, 0, stream>>>(fb, W2t, h2, N);
  selfloop_bias_kernel<<<gND, B, 0, stream>>>(h2, dis, b2, x1o, ND);
  spmm_edge_kernel<<<gE, B, 0, stream>>>(h2, ew, row, col, dis, x1o, E);

  // ---- euclidean branch, layer 1 ----
  fill_kernel<<<gN, B, 0, stream>>>(deg, 1.0f, N);
  edge_euc_kernel<<<gE, B, 0, stream>>>(x, row, col, ew, deg, E);
  dis_kernel<<<gN, B, 0, stream>>>(deg, dis, N);
  selfloop_bias_kernel<<<gND, B, 0, stream>>>(h1, dis, b1, c2, ND);
  spmm_edge_kernel<<<gE, B, 0, stream>>>(h1, ew, row, col, dis, c2, E);
  relu_kernel<<<gND, B, 0, stream>>>(c2, ND);
  // ---- euclidean branch, layer 2 ----
  fill_kernel<<<gN, B, 0, stream>>>(deg, 1.0f, N);
  edge_euc_kernel<<<gE, B, 0, stream>>>(c2, row, col, ew, deg, E);
  dis_kernel<<<gN, B, 0, stream>>>(deg, dis, N);
  convert_bf16_kernel<<<gND, B, 0, stream>>>(c2, fb, ND);
  gemm_bf16_wmma<<<gM, B, 0, stream>>>(fb, W2t, h2, N);
  selfloop_bias_kernel<<<gND, B, 0, stream>>>(h2, dis, b2, x2o, ND);
  spmm_edge_kernel<<<gE, B, 0, stream>>>(h2, ew, row, col, dis, x2o, E);

  // ---- semantic attention fusion ----
  fusion_kernel<<<gF, B, 0, stream>>>(x1o, x2o, A1, ab1, A2, out, N);
}
